// NTXentLoss_27779848471490
// MI455X (gfx1250) — matrix operand order repaired and checked by reading, not compiled, not run on hardware
//
#include <hip/hip_runtime.h>
#include <hip/hip_bf16.h>
#include <math.h>

// ---------------------------------------------------------------------------
// NT-Xent loss, fused flash-style on gfx1250 (MI455X, wave32, WMMA).
//   B = 8192, D = 128, N = 2B = 16384, T = 0.5, tau+ = 0.1
// Heavy op: 16384x16384x128 similarity GEMM (68.7 GFLOP) fused with a
// streaming masked exp-sum so the 1 GB sim matrix never touches memory.
// Column dim split into CHUNKS for occupancy; fixed-shift exp partials
// (s = sum exp(sim/T - 2), m' = rowmax - 2) combine exactly and
// deterministically. Diagonal is accumulated unconditionally in the hot loop
// and subtracted analytically afterwards (no divergent branch in the loop).
// ---------------------------------------------------------------------------

typedef __attribute__((ext_vector_type(16))) _Float16 v16h;
typedef __attribute__((ext_vector_type(8)))  _Float16 v8h;
typedef __attribute__((ext_vector_type(4)))  _Float16 v4h;
typedef __attribute__((ext_vector_type(8)))  float    v8f;

#define NTX_B    8192
#define NTX_D    128
#define NTX_N    16384   // 2B
#define NTX_INVT 2.0f    // 1/TEMPERATURE
#define NTX_CHUNKS 4     // column-dimension split (occupancy)

// ---------------------------------------------------------------------------
// Kernel 1: L2-normalize z_i / z_j (fp32), write f16 reps, exact fp32 pos dot.
// One wave per row-pair; lane l owns elements [4l, 4l+4).
// ---------------------------------------------------------------------------
__global__ __launch_bounds__(256) void ntx_normalize(
    const float* __restrict__ z_i, const float* __restrict__ z_j,
    _Float16* __restrict__ repsH, float* __restrict__ pos)
{
    const int wave = threadIdx.x >> 5;
    const int lane = threadIdx.x & 31;
    const int r = blockIdx.x * 8 + wave;

    const float4 vi = reinterpret_cast<const float4*>(z_i + (size_t)r * NTX_D)[lane];
    const float4 vj = reinterpret_cast<const float4*>(z_j + (size_t)r * NTX_D)[lane];

    float ssi = vi.x*vi.x + vi.y*vi.y + vi.z*vi.z + vi.w*vi.w;
    float ssj = vj.x*vj.x + vj.y*vj.y + vj.z*vj.z + vj.w*vj.w;
    float cr  = vi.x*vj.x + vi.y*vj.y + vi.z*vj.z + vi.w*vj.w;

#pragma unroll
    for (int off = 16; off > 0; off >>= 1) {
        ssi += __shfl_xor(ssi, off, 32);
        ssj += __shfl_xor(ssj, off, 32);
        cr  += __shfl_xor(cr,  off, 32);
    }

    const float ni = fmaxf(sqrtf(ssi), 1e-12f);   // matches F.normalize clamp
    const float nj = fmaxf(sqrtf(ssj), 1e-12f);
    const float inv_i = 1.0f / ni, inv_j = 1.0f / nj;

    v4h hi, hj;
    hi[0] = (_Float16)(vi.x * inv_i); hi[1] = (_Float16)(vi.y * inv_i);
    hi[2] = (_Float16)(vi.z * inv_i); hi[3] = (_Float16)(vi.w * inv_i);
    hj[0] = (_Float16)(vj.x * inv_j); hj[1] = (_Float16)(vj.y * inv_j);
    hj[2] = (_Float16)(vj.z * inv_j); hj[3] = (_Float16)(vj.w * inv_j);

    *reinterpret_cast<v4h*>(repsH + (size_t)r * NTX_D + lane * 4)            = hi;
    *reinterpret_cast<v4h*>(repsH + (size_t)(r + NTX_B) * NTX_D + lane * 4)  = hj;

    if (lane == 0) {
        const float dotn = cr * inv_i * inv_j;            // exact fp32 positive dot
        pos[r] = __expf(NTX_INVT * dotn);                 // exp(sim(zi,zj)/T)
    }
}

// ---------------------------------------------------------------------------
// Fragment loaders for v_wmma_f32_16x16x32_f16 (CDNA5 ISA 7.12.2):
// A (16Mx32K):  lanes 0-15 : M=lane,    halves = K kbase+[0..15]   (contiguous; code's own k order)
//               lanes16-31 : M=lane-16, halves = K kbase+[16..31]  (contiguous; same as B)
// B (32Kx16N):  lanes 0-15 : N=lane,    halves = K kbase+[0..15]   (contiguous)
//               lanes16-31 : N=lane-16, halves = K kbase+[16..31]  (contiguous)
// For sim = reps . reps^T, B[k][n] = reps[c0+n][k]: per-lane B chunk is 32
// contiguous bytes of a reps row.
// ---------------------------------------------------------------------------
__device__ __forceinline__ v16h ntx_load_afrag(const _Float16* __restrict__ rows,
                                               int lane)
{
    const int ln = lane & 15;
    const int hf = lane >> 4;
    const _Float16* p = rows + (size_t)ln * NTX_D + hf * 16;
    const v8h c0 = *reinterpret_cast<const v8h*>(p);        // K +{0|16}..7
    const v8h c1 = *reinterpret_cast<const v8h*>(p + 8);    // K +{8|24}..7
    v16h r;
#pragma unroll
    for (int i = 0; i < 8; ++i) { r[i] = c0[i]; r[i + 8] = c1[i]; }
    return r;
}

__device__ __forceinline__ v16h ntx_load_bfrag(const _Float16* __restrict__ rows,
                                               int lane)
{
    const int ln = lane & 15;
    const int hf = lane >> 4;
    const _Float16* p = rows + (size_t)ln * NTX_D + hf * 16; // 32B contiguous
    const v8h c0 = *reinterpret_cast<const v8h*>(p);
    const v8h c1 = *reinterpret_cast<const v8h*>(p + 8);
    v16h r;
#pragma unroll
    for (int i = 0; i < 8; ++i) { r[i] = c0[i]; r[i + 8] = c1[i]; }
    return r;
}

// ---------------------------------------------------------------------------
// Kernel 2: fused sim-row pass over one column chunk.
//   grid = (N/64 row blocks, NTX_CHUNKS column chunks), 4 waves/WG,
//   each wave owns 16 rows x (N/CHUNKS) columns -> row state stays in-wave.
// Hot loop per tile: 8x b128 loads for tile t+1 (pipelined), 4 chained WMMAs,
// then per element y = 2*acc - 2:  mt = max(mt,y); s += exp(y). No branches.
// The diagonal (present in exactly one tile of one chunk) is included, then
// subtracted after the loop using x_diag = 2*||row||^2_f16 recomputed from the
// resident A-fragments (order-only difference vs the WMMA diag, ~1 ulp).
// ---------------------------------------------------------------------------
__global__ __launch_bounds__(128) void ntx_simrow(
    const _Float16* __restrict__ repsH,
    float* __restrict__ spart, float* __restrict__ mpart)
{
    const int wave = threadIdx.x >> 5;
    const int lane = threadIdx.x & 31;
    const int ln   = lane & 15;
    const int hf   = lane >> 4;
    const int row0 = blockIdx.x * 64 + wave * 16;
    const int myrow0 = row0 + hf * 8;        // this lane's rows: myrow0 + v

    constexpr int NT = NTX_N / 16;           // 1024 column tiles total
    constexpr int CT = NT / NTX_CHUNKS;      // tiles per chunk
    const int tstart = blockIdx.y * CT;
    const int tend   = tstart + CT;
    // chunk containing all 4 waves' diagonal tiles (block-uniform)
    const bool diag_chunk = ((int)blockIdx.y == ((int)blockIdx.x >> 6));

    const _Float16* arow = repsH + (size_t)row0 * NTX_D;
    const v16h a0 = ntx_load_afrag(arow +  0, lane);
    const v16h a1 = ntx_load_afrag(arow + 32, lane);
    const v16h a2 = ntx_load_afrag(arow + 64, lane);
    const v16h a3 = ntx_load_afrag(arow + 96, lane);

    float mt[8], s[8];
#pragma unroll
    for (int v = 0; v < 8; ++v) { mt[v] = -__builtin_inff(); s[v] = 0.0f; }

    // prime the pipeline with the chunk's first tile
    v16h b0 = ntx_load_bfrag(repsH + (size_t)tstart * 16 * NTX_D +  0, lane);
    v16h b1 = ntx_load_bfrag(repsH + (size_t)tstart * 16 * NTX_D + 32, lane);
    v16h b2 = ntx_load_bfrag(repsH + (size_t)tstart * 16 * NTX_D + 64, lane);
    v16h b3 = ntx_load_bfrag(repsH + (size_t)tstart * 16 * NTX_D + 96, lane);

    for (int t = tstart; t < tend; ++t) {
        // issue tile t+1 loads now (wraps to chunk start on last iter; discarded)
        const int tn = (t + 1 < tend) ? (t + 1) : tstart;
        const int tp = (t + 2 < tend) ? (t + 2) : tstart;
        const _Float16* nrow = repsH + (size_t)tn * 16 * NTX_D;
        __builtin_prefetch(repsH + (size_t)tp * 16 * NTX_D + lane * 64, 0, 3);
        const v16h c0 = ntx_load_bfrag(nrow +  0, lane);
        const v16h c1 = ntx_load_bfrag(nrow + 32, lane);
        const v16h c2 = ntx_load_bfrag(nrow + 64, lane);
        const v16h c3 = ntx_load_bfrag(nrow + 96, lane);

        v8f acc = {};
        acc = __builtin_amdgcn_wmma_f32_16x16x32_f16(false, a0, false, b0,
                                                     (short)0, acc, false, false);
        acc = __builtin_amdgcn_wmma_f32_16x16x32_f16(false, a1, false, b1,
                                                     (short)0, acc, false, false);
        acc = __builtin_amdgcn_wmma_f32_16x16x32_f16(false, a2, false, b2,
                                                     (short)0, acc, false, false);
        acc = __builtin_amdgcn_wmma_f32_16x16x32_f16(false, a3, false, b3,
                                                     (short)0, acc, false, false);

#pragma unroll
        for (int v = 0; v < 8; ++v) {                    // branch-free epilogue
            const float y = __builtin_fmaf(NTX_INVT, acc[v], -2.0f); // sim/T - 2
            mt[v] = fmaxf(mt[v], y);                     // rowmax' (diag included)
            s[v] += __expf(y);                           // never overflows (y<~0.01)
        }

        b0 = c0; b1 = c1; b2 = c2; b3 = c3;
    }

    // Recompute this wave's diagonal values from the resident A-fragments:
    // lane l and l+16 together hold row row0 + (l%16); one xor-16 combines.
    float pq = 0.0f;
#pragma unroll
    for (int i = 0; i < 16; ++i) {
        pq = __builtin_fmaf((float)a0[i], (float)a0[i], pq);
        pq = __builtin_fmaf((float)a1[i], (float)a1[i], pq);
        pq = __builtin_fmaf((float)a2[i], (float)a2[i], pq);
        pq = __builtin_fmaf((float)a3[i], (float)a3[i], pq);
    }
    pq += __shfl_xor(pq, 16, 32);            // ||row(row0+ln)||^2 at every lane

    // Combine the 16 lanes of each half-wave (same rows, disjoint columns),
    // subtract the diagonal term where applicable, write chunk partials.
    float* sdst = spart + (size_t)blockIdx.y * NTX_N;
    float* mdst = mpart + (size_t)blockIdx.y * NTX_N;
#pragma unroll
    for (int v = 0; v < 8; ++v) {
        float mm = mt[v], ss = s[v];
#pragma unroll
        for (int off = 1; off < 16; off <<= 1) {         // stays within 16-lane half
            mm = fmaxf(mm, __shfl_xor(mm, off, 32));
            ss += __shfl_xor(ss, off, 32);
        }
        // rowsq of row myrow0+v lives at absolute lane hf*8+v
        const float rq = __shfl(pq, hf * 8 + v, 32);
        if (diag_chunk)
            ss -= __expf(__builtin_fmaf(2.0f, rq, -2.0f));   // remove exp(x_diag-2)
        if (ln == 0) { sdst[myrow0 + v] = ss; mdst[myrow0 + v] = mm; }
    }
}

// ---------------------------------------------------------------------------
// Kernel 3: fold chunk partials -> negsum -> mean loss (one block, exact order,
// bitwise deterministic across replays). mpart holds rowmax-2, so
// negsum = (sum exp(sim/T - 2)) * exp(-(rowmax-2)) = sum exp(sim - rowmax).
// ---------------------------------------------------------------------------
__global__ __launch_bounds__(256) void ntx_loss(
    const float* __restrict__ pos,
    const float* __restrict__ spart, const float* __restrict__ mpart,
    float* __restrict__ out)
{
    __shared__ float red[8];
    const float ng_floor = (float)NTX_B * __expf(-NTX_INVT);   // B*exp(-1/T)
    float acc = 0.0f;
    for (int r = threadIdx.x; r < NTX_N; r += 256) {
        float st = 0.0f, mm = -__builtin_inff();
#pragma unroll
        for (int c = 0; c < NTX_CHUNKS; ++c) {
            st += spart[(size_t)c * NTX_N + r];
            mm = fmaxf(mm, mpart[(size_t)c * NTX_N + r]);
        }
        const float ns = st * __expf(-mm);    // = sum_{j!=i} exp(sim - rowmax)
        const float p  = pos[r & (NTX_B - 1)];
        float Ng = (ns - 0.1f * (float)NTX_B * p) * (1.0f / 0.9f);
        Ng = fmaxf(Ng, ng_floor);
        acc += -__logf(p / (p + Ng + 1e-8f));
    }
#pragma unroll
    for (int off = 16; off > 0; off >>= 1) acc += __shfl_xor(acc, off, 32);
    if ((threadIdx.x & 31) == 0) red[threadIdx.x >> 5] = acc;
    __syncthreads();
    if (threadIdx.x < 8) {
        float a = red[threadIdx.x];
#pragma unroll
        for (int off = 4; off > 0; off >>= 1) a += __shfl_xor(a, off, 32);
        if (threadIdx.x == 0) out[0] = a * (1.0f / (float)NTX_N);
    }
}

// ---------------------------------------------------------------------------
extern "C" void kernel_launch(void* const* d_in, const int* in_sizes, int n_in,
                              void* d_out, int out_size, void* d_ws, size_t ws_size,
                              hipStream_t stream) {
    const float* z_i = (const float*)d_in[0];
    const float* z_j = (const float*)d_in[1];
    float* out = (float*)d_out;

    char* ws = (char*)d_ws;
    _Float16* repsH = (_Float16*)ws;                                    // 4 MB
    size_t off = (size_t)NTX_N * NTX_D * 2;
    float* pos   = (float*)(ws + off);  off += (size_t)NTX_B * 4;       // 32 KB
    float* spart = (float*)(ws + off);  off += (size_t)NTX_CHUNKS * NTX_N * 4;
    float* mpart = (float*)(ws + off);                                  // 256 KB each

    ntx_normalize<<<NTX_B / 8, 256, 0, stream>>>(z_i, z_j, repsH, pos);
    ntx_simrow<<<dim3(NTX_N / 64, NTX_CHUNKS), 128, 0, stream>>>(repsH, spart, mpart);
    ntx_loss<<<1, 256, 0, stream>>>(pos, spart, mpart, out);
}
